// PNAConv_84009560309791
// MI455X (gfx1250) — matrix operand order repaired and checked by reading, not compiled, not run on hardware
//
#include <hip/hip_runtime.h>
#include <hip/hip_bf16.h>
#include <math.h>

// ---------------------------------------------------------------------------
// PNAConv on MI455X (gfx1250), wave32, bf16 WMMA path, CSR aggregation.
// ---------------------------------------------------------------------------

typedef __attribute__((ext_vector_type(16))) __bf16 v16bf;
typedef __attribute__((ext_vector_type(8)))  __bf16 v8bf;
typedef __attribute__((ext_vector_type(8)))  float  v8f;

#define NN   50000
#define EE   800000
#define FF   128
#define KMSG 384          // [x_dst | x_src | edge_attr] folded K
#define KPST 1664         // [x | 12F scaled] folded K
#define AVG_LOG 1.6904263f

// LDS row padding (bf16 elements) to spread rows across the 64 banks.
#define PAD_MSG  392      // 384 + 8  -> 784B row stride (196 dwords, %64 = 4)
#define PAD_PST  1672     // 1664 + 8 -> 3344B row stride (836 dwords, %64 = 4)

// ---------------------------------------------------------------------------
// Prep: W_ep = W_edge @ W_pre[256:384,:]   (fp32)
// ---------------------------------------------------------------------------
__global__ void k_prep_wep(const float* __restrict__ W_edge,
                           const float* __restrict__ W_pre,
                           float* __restrict__ W_ep) {
    int idx = blockIdx.x * blockDim.x + threadIdx.x;
    if (idx >= FF * FF) return;
    int k = idx >> 7, n = idx & 127;
    float s = 0.f;
    for (int j = 0; j < FF; ++j)
        s = fmaf(W_edge[k * FF + j], W_pre[(256 + j) * FF + n], s);
    W_ep[idx] = s;
}

// Transposed bf16 message weight Wm_t[n][k] + fused bias bm
__global__ void k_build_wm(const float* __restrict__ W_pre,
                           const float* __restrict__ W_ep,
                           const float* __restrict__ b_pre,
                           const float* __restrict__ b_edge,
                           __bf16* __restrict__ Wm_t,
                           float* __restrict__ bm) {
    int idx = blockIdx.x * blockDim.x + threadIdx.x;
    if (idx < FF * KMSG) {
        int n = idx / KMSG, k = idx - n * KMSG;
        float v = (k < 256) ? W_pre[k * FF + n] : W_ep[(k - 256) * FF + n];
        Wm_t[(size_t)n * KMSG + k] = (__bf16)v;
    }
    if (idx < FF) {
        float s = b_pre[idx];
        for (int j = 0; j < FF; ++j)
            s = fmaf(b_edge[j], W_pre[(256 + j) * FF + idx], s);
        bm[idx] = s;
    }
}

// W_pl = W_post @ W_lin  (fp32 [1664,128])
__global__ void k_prep_wpl(const float* __restrict__ W_post,
                           const float* __restrict__ W_lin,
                           float* __restrict__ W_pl) {
    int idx = blockIdx.x * blockDim.x + threadIdx.x;
    if (idx >= KPST * FF) return;
    int k = idx >> 7, n = idx & 127;
    float s = 0.f;
    for (int j = 0; j < FF; ++j)
        s = fmaf(W_post[k * FF + j], W_lin[j * FF + n], s);
    W_pl[idx] = s;
}

// Transposed bf16 W_pl_t[n][k] + fused bias bpl
__global__ void k_build_wpl(const float* __restrict__ W_pl,
                            const float* __restrict__ b_post,
                            const float* __restrict__ b_lin,
                            const float* __restrict__ W_lin,
                            __bf16* __restrict__ Wpl_t,
                            float* __restrict__ bpl) {
    int idx = blockIdx.x * blockDim.x + threadIdx.x;
    if (idx < FF * KPST) {
        int n = idx / KPST, k = idx - n * KPST;
        Wpl_t[(size_t)n * KPST + k] = (__bf16)W_pl[k * FF + n];
    }
    if (idx < FF) {
        float s = b_lin[idx];
        for (int j = 0; j < FF; ++j)
            s = fmaf(b_post[j], W_lin[j * FF + idx], s);
        bpl[idx] = s;
    }
}

// ---------------------------------------------------------------------------
// CSR build: degree histogram -> exclusive scan -> bucket scatter
// ---------------------------------------------------------------------------
__global__ void k_init_csr(int* __restrict__ deg, int* __restrict__ cursor) {
    int i = blockIdx.x * blockDim.x + threadIdx.x;
    if (i < NN) { deg[i] = 0; cursor[i] = 0; }
}

__global__ void k_hist(const int* __restrict__ dst, int* __restrict__ deg) {
    int e = blockIdx.x * blockDim.x + threadIdx.x;
    if (e < EE) atomicAdd(&deg[dst[e]], 1);
}

// Single-block Hillis-Steele scan over NN entries (exclusive)
__global__ void __launch_bounds__(256, 1)
k_scan(const int* __restrict__ deg, int* __restrict__ rowstart) {
    __shared__ int buf[256];
    __shared__ int carry;
    const int t = threadIdx.x;
    if (t == 0) carry = 0;
    __syncthreads();
    for (int base = 0; base < NN; base += 256) {
        const int i = base + t;
        const int v = (i < NN) ? deg[i] : 0;
        buf[t] = v;
        __syncthreads();
        #pragma unroll
        for (int ofs = 1; ofs < 256; ofs <<= 1) {
            int add = (t >= ofs) ? buf[t - ofs] : 0;
            __syncthreads();
            buf[t] += add;
            __syncthreads();
        }
        if (i < NN) rowstart[i] = carry + buf[t] - v;   // exclusive
        __syncthreads();
        if (t == 0) carry += buf[255];
        __syncthreads();
    }
    if (t == 0) rowstart[NN] = carry;
}

__global__ void k_scatter(const int* __restrict__ dst,
                          const int* __restrict__ rowstart,
                          int* __restrict__ cursor,
                          int* __restrict__ eord) {
    int e = blockIdx.x * blockDim.x + threadIdx.x;
    if (e >= EE) return;
    const int d = dst[e];
    const int pos = rowstart[d] + atomicAdd(&cursor[d], 1);
    eord[pos] = e;
}

// ---------------------------------------------------------------------------
// Message kernel: m = [x_dst | x_src | edge_attr] @ Wm + bm  (bf16 WMMA).
// Block = 32 edges (two M-tiles), 8 waves; each B fragment feeds 2 WMMAs.
// Messages stored fp16 to mbuf[E,128].
// ---------------------------------------------------------------------------
__global__ void __launch_bounds__(256, 1)
k_msg(const float* __restrict__ x, const float* __restrict__ ea,
      const int* __restrict__ src, const int* __restrict__ dst,
      const __bf16* __restrict__ Wm_t, const float* __restrict__ bm,
      _Float16* __restrict__ mbuf) {
    __shared__ __align__(16) __bf16 ldsA[32 * PAD_MSG];
    __shared__ int ldsDst[32];
    __shared__ int ldsSrc[32];

    const int t  = threadIdx.x;
    const size_t e0 = (size_t)blockIdx.x * 32;

    if (t < 32)               ldsDst[t]      = dst[e0 + t];
    else if (t < 64)          ldsSrc[t - 32] = src[e0 + t - 32];
    __syncthreads();

    // Stage A (32 edges x 384) into LDS as bf16 (coalesced fp32 reads)
    for (int i = t; i < 32 * KMSG; i += 256) {
        int m = i / KMSG, k = i - m * KMSG;
        float v;
        if (k < 128)      v = x[(size_t)ldsDst[m] * FF + k];
        else if (k < 256) v = x[(size_t)ldsSrc[m] * FF + (k - 128)];
        else              v = ea[(e0 + (size_t)m) * FF + (k - 256)];
        ldsA[m * PAD_MSG + k] = (__bf16)v;
    }
    __syncthreads();

    const int wave = t >> 5, lane = t & 31;
    const int col  = lane & 15, half = lane >> 4;
    const int ncol = wave * 16 + col;

    v8f acc0 = {}, acc1 = {};
    // A fragments per ISA 7.12.2: lane M = lane&15; half selects K 0..7/8..15 (+16..23/24..31)
    const __bf16* Ap0 = ldsA + (lane & 15) * PAD_MSG + half * 8;
    const __bf16* Ap1 = Ap0 + 16 * PAD_MSG;
    // B fragment: column n = ncol; half selects K 0..7/8..15 (+16..23/24..31)
    const __bf16* Bp = Wm_t + (size_t)ncol * KMSG + half * 8;

    #pragma unroll
    for (int kb = 0; kb < 12; ++kb) {
        const int koff = kb * 32;
        __builtin_prefetch(Bp + koff + 64, 0, 1);   // global_prefetch_b8
        v16bf b;
        {
            const v8bf* p = (const v8bf*)(Bp + koff);
            v8bf lo = p[0], hi = p[2];
            #pragma unroll
            for (int i = 0; i < 8; ++i) { b[i] = lo[i]; b[8 + i] = hi[i]; }
        }
        v16bf a0, a1;
        {
            v8bf lo = *(const v8bf*)(Ap0 + koff);
            v8bf hi = *(const v8bf*)(Ap0 + koff + 16);
            #pragma unroll
            for (int i = 0; i < 8; ++i) { a0[i] = lo[i]; a0[8 + i] = hi[i]; }
        }
        {
            v8bf lo = *(const v8bf*)(Ap1 + koff);
            v8bf hi = *(const v8bf*)(Ap1 + koff + 16);
            #pragma unroll
            for (int i = 0; i < 8; ++i) { a1[i] = lo[i]; a1[8 + i] = hi[i]; }
        }
        acc0 = __builtin_amdgcn_wmma_f32_16x16x32_bf16(
                   false, a0, false, b, (short)0, acc0, false, false);
        acc1 = __builtin_amdgcn_wmma_f32_16x16x32_bf16(
                   false, a1, false, b, (short)0, acc1, false, false);
    }

    const float bias = bm[ncol];
    #pragma unroll
    for (int r = 0; r < 8; ++r) {
        const int row = r + half * 8;           // C layout: half selects M 0-7 / 8-15
        mbuf[(e0 + row)      * FF + ncol] = (_Float16)(acc0[r] + bias);
        mbuf[(e0 + 16 + row) * FF + ncol] = (_Float16)(acc1[r] + bias);
    }
}

// ---------------------------------------------------------------------------
// Fused CSR aggregation + PNA scalers -> bf16 post-activation rows.
// Block = 256 threads = 2 nodes x 128 features; contiguous 256B row reads.
// ---------------------------------------------------------------------------
__global__ void __launch_bounds__(256, 1)
k_aggr(const float* __restrict__ x,
       const int* __restrict__ rowstart,
       const int* __restrict__ eord,
       const _Float16* __restrict__ mbuf,
       __bf16* __restrict__ Apost) {
    const int n = blockIdx.x * 2 + (threadIdx.x >> 7);
    const int f = threadIdx.x & 127;
    if (n >= NN) return;

    const int r0 = rowstart[n], r1 = rowstart[n + 1];
    float s = 0.f, s2 = 0.f, vmn = INFINITY, vmx = -INFINITY;
    for (int j = r0; j < r1; ++j) {
        const float v = (float)mbuf[(size_t)eord[j] * FF + f];
        s += v; s2 += v * v;
        vmn = fminf(vmn, v); vmx = fmaxf(vmx, v);
    }
    const float c  = (float)(r1 - r0);
    const float cc = fmaxf(c, 1.f);
    const float mean = s / cc;
    const float var  = s2 / cc - mean * mean;
    const float sd   = sqrtf(fmaxf(var, 0.f) + 1e-5f);
    const bool  has  = c > 0.f;
    if (!has) { vmn = 0.f; vmx = 0.f; }

    const float ld  = logf(cc + 1.f);
    const float amp = ld / AVG_LOG;
    const float att = AVG_LOG / ld;

    __bf16* row = Apost + (size_t)n * KPST;
    row[f] = (__bf16)x[(size_t)n * FF + f];
    const float a0 = mean, a1 = vmn, a2 = vmx, a3 = sd;
    row[ 128 + f] = (__bf16)a0;          row[ 256 + f] = (__bf16)a1;
    row[ 384 + f] = (__bf16)a2;          row[ 512 + f] = (__bf16)a3;
    row[ 640 + f] = (__bf16)(a0 * amp);  row[ 768 + f] = (__bf16)(a1 * amp);
    row[ 896 + f] = (__bf16)(a2 * amp);  row[1024 + f] = (__bf16)(a3 * amp);
    row[1152 + f] = (__bf16)(a0 * att);  row[1280 + f] = (__bf16)(a1 * att);
    row[1408 + f] = (__bf16)(a2 * att);  row[1536 + f] = (__bf16)(a3 * att);
}

// ---------------------------------------------------------------------------
// Post GEMM: out[N,128] = Apost[N,1664] @ W_pl + bpl   (bf16 WMMA, fp32 out)
// ---------------------------------------------------------------------------
__global__ void __launch_bounds__(256, 1)
k_post(const __bf16* __restrict__ Apost, const __bf16* __restrict__ Wpl_t,
       const float* __restrict__ bpl, float* __restrict__ out) {
    __shared__ __align__(16) __bf16 ldsA[16 * PAD_PST];
    const int t  = threadIdx.x;
    const size_t n0 = (size_t)blockIdx.x * 16;

    for (int c = t; c < 16 * 208; c += 256) {
        const int m  = c / 208;
        const int kk = (c - m * 208) * 8;
        *(v8bf*)(ldsA + m * PAD_PST + kk) =
            *(const v8bf*)(Apost + (n0 + (size_t)m) * KPST + kk);
    }
    __syncthreads();

    const int wave = t >> 5, lane = t & 31;
    const int col  = lane & 15, half = lane >> 4;
    const int ncol = wave * 16 + col;

    v8f acc = {};
    const __bf16* Ap = ldsA + (lane & 15) * PAD_PST + half * 8;
    const __bf16* Bp = Wpl_t + (size_t)ncol * KPST + half * 8;

    #pragma unroll 4
    for (int kb = 0; kb < 52; ++kb) {
        const int koff = kb * 32;
        __builtin_prefetch(Bp + koff + 64, 0, 1);
        v16bf a, b;
        {
            v8bf lo = *(const v8bf*)(Ap + koff);
            v8bf hi = *(const v8bf*)(Ap + koff + 16);
            #pragma unroll
            for (int i = 0; i < 8; ++i) { a[i] = lo[i]; a[8 + i] = hi[i]; }
        }
        {
            const v8bf* p = (const v8bf*)(Bp + koff);
            v8bf lo = p[0], hi = p[2];
            #pragma unroll
            for (int i = 0; i < 8; ++i) { b[i] = lo[i]; b[8 + i] = hi[i]; }
        }
        acc = __builtin_amdgcn_wmma_f32_16x16x32_bf16(
                  false, a, false, b, (short)0, acc, false, false);
    }

    const float bias = bpl[ncol];
    #pragma unroll
    for (int r = 0; r < 8; ++r) {
        const int row = r + half * 8;
        out[(n0 + (size_t)row) * FF + ncol] = acc[r] + bias;
    }
}

// ---------------------------------------------------------------------------
// Host-side launch
// ---------------------------------------------------------------------------
static inline size_t align256(size_t v) { return (v + 255) & ~(size_t)255; }

extern "C" void kernel_launch(void* const* d_in, const int* in_sizes, int n_in,
                              void* d_out, int out_size, void* d_ws, size_t ws_size,
                              hipStream_t stream) {
    const float* x      = (const float*)d_in[0];
    const float* ea     = (const float*)d_in[1];
    const float* W_edge = (const float*)d_in[2];
    const float* b_edge = (const float*)d_in[3];
    const float* W_pre  = (const float*)d_in[4];
    const float* b_pre  = (const float*)d_in[5];
    const float* W_post = (const float*)d_in[6];
    const float* b_post = (const float*)d_in[7];
    const float* W_lin  = (const float*)d_in[8];
    const float* b_lin  = (const float*)d_in[9];
    const int*   eidx   = (const int*)  d_in[10];
    const int*   srcI   = eidx;          // edge_index[0]
    const int*   dstI   = eidx + EE;     // edge_index[1]
    float*       out    = (float*)d_out;

    // Carve workspace
    char* p = (char*)d_ws;
    size_t off = 0;
    __bf16*    Wm_t   = (__bf16*)   (p + off); off = align256(off + (size_t)FF * KMSG * 2);
    float*     bm     = (float*)    (p + off); off = align256(off + FF * 4);
    __bf16*    Wpl_t  = (__bf16*)   (p + off); off = align256(off + (size_t)FF * KPST * 2);
    float*     bpl    = (float*)    (p + off); off = align256(off + FF * 4);
    float*     W_ep   = (float*)    (p + off); off = align256(off + (size_t)FF * FF * 4);
    float*     W_pl   = (float*)    (p + off); off = align256(off + (size_t)KPST * FF * 4);
    int*       deg    = (int*)      (p + off); off = align256(off + (size_t)NN * 4);
    int*       rowstart=(int*)      (p + off); off = align256(off + (size_t)(NN + 1) * 4);
    int*       cursor = (int*)      (p + off); off = align256(off + (size_t)NN * 4);
    int*       eord   = (int*)      (p + off); off = align256(off + (size_t)EE * 4);
    _Float16*  mbuf   = (_Float16*) (p + off); off = align256(off + (size_t)EE * FF * 2);
    __bf16*    Apost  = (__bf16*)   (p + off); off = align256(off + (size_t)NN * KPST * 2);
    (void)ws_size; (void)in_sizes; (void)n_in; (void)out_size;

    // Weight folding / bf16 packing
    k_prep_wep <<<(FF * FF + 255) / 256, 256, 0, stream>>>(W_edge, W_pre, W_ep);
    k_build_wm <<<(FF * KMSG + 255) / 256, 256, 0, stream>>>(W_pre, W_ep, b_pre, b_edge, Wm_t, bm);
    k_prep_wpl <<<(KPST * FF + 255) / 256, 256, 0, stream>>>(W_post, W_lin, W_pl);
    k_build_wpl<<<(FF * KPST + 255) / 256, 256, 0, stream>>>(W_pl, b_post, b_lin, W_lin, Wpl_t, bpl);

    // CSR build
    k_init_csr<<<(NN + 255) / 256, 256, 0, stream>>>(deg, cursor);
    k_hist    <<<(EE + 255) / 256, 256, 0, stream>>>(dstI, deg);
    k_scan    <<<1, 256, 0, stream>>>(deg, rowstart);
    k_scatter <<<(EE + 255) / 256, 256, 0, stream>>>(dstI, rowstart, cursor, eord);

    // Edge messages (bf16 WMMA, 32 edges/block, B reused across 2 M-tiles)
    k_msg<<<EE / 32, 256, 0, stream>>>(x, ea, srcI, dstI, Wm_t, bm, mbuf);

    // CSR aggregation + PNA scalers (fused) -> bf16 activation matrix
    k_aggr<<<(NN + 1) / 2, 256, 0, stream>>>(x, rowstart, eord, mbuf, Apost);

    // Fused post-MLP + linear (bf16 WMMA)
    k_post<<<NN / 16, 256, 0, stream>>>(Apost, Wpl_t, bpl, out);
}
